// Block_42159398977962
// MI455X (gfx1250) — matrix operand order repaired and checked by reading, not compiled, not run on hardware
//
#include <hip/hip_runtime.h>

typedef __bf16 bf16;
typedef __attribute__((ext_vector_type(8)))  bf16  v8bf;
typedef __attribute__((ext_vector_type(16))) bf16  v16bf;
typedef __attribute__((ext_vector_type(8)))  float v8f;

constexpr int kB = 4, kT = 2048, kC = 1024, kH = 16, kD = 64, kF = 4096;
constexpr int kBT = kB * kT;               // 8192 rows
constexpr float kScale = 1.0f / 4096.0f;   // faithful: 1/DQK^2

// ---------------------------------------------------------------- helpers ---

__device__ __forceinline__ bf16 to_bf16(float f) {
  unsigned int u = __builtin_bit_cast(unsigned int, f);
  u += 0x7FFFu + ((u >> 16) & 1u);          // round-to-nearest-even
  return __builtin_bit_cast(bf16, (unsigned short)(u >> 16));
}

__device__ __forceinline__ v16bf load16(const bf16* p0, const bf16* p1) {
  v8bf lo = *(const v8bf*)p0;
  v8bf hi = *(const v8bf*)p1;
  return __builtin_shufflevector(lo, hi, 0,1,2,3,4,5,6,7,8,9,10,11,12,13,14,15);
}

__device__ __forceinline__ v8f wmma_bf16(v16bf a, v16bf b, v8f c) {
  return __builtin_amdgcn_wmma_f32_16x16x32_bf16(
      false, a, false, b, (short)0, c, false, false);
}

// A fragment (16x32 bf16 tile rows):
//   lane<16: K chunks at k0+0..7 / k0+16..23 ; lane>=16: k0+8..15 / k0+24..31
__device__ __forceinline__ v16bf load_a(const bf16* row, int k0, int lane) {
  int sa = (lane & 16) ? 8 : 0;
  return load16(row + k0 + sa, row + k0 + sa + 16);
}

// B fragment from [N][K] row-major (B transposed):
//   lane<16: K chunks at k0+0..7 / k0+16..23 ; lane>=16: k0+8..15 / k0+24..31
__device__ __forceinline__ v16bf load_b(const bf16* row, int k0, int lane) {
  int sb = (lane & 16) ? 8 : 0;
  return load16(row + k0 + sb, row + k0 + sb + 16);
}

// gfx1250 async global->LDS copy, 16 bytes per lane (ASYNCcnt-tracked)
__device__ __forceinline__ void async_b128(unsigned lds_off, const void* base,
                                           unsigned voff) {
  asm volatile("global_load_async_to_lds_b128 %0, %1, %2"
               :: "v"(lds_off), "v"(voff), "s"(base) : "memory");
}
__device__ __forceinline__ void wait_async0() {
  asm volatile("s_wait_asynccnt 0x0" ::: "memory");
}
__device__ __forceinline__ unsigned lds_addr(const void* p) {
  return (unsigned)(uintptr_t)p;   // low 32 bits of the generic address used as LDS address
}

// ----------------------------------------------------- weight transpose -----
__global__ __launch_bounds__(256) void cvt_t_kernel(
    const float* __restrict__ W, bf16* __restrict__ WT, int R, int Ncol) {
  size_t z = blockIdx.y;
  size_t total = (size_t)R * Ncol;
  const float* in = W + z * total;
  bf16* out = WT + z * total;
  size_t i = (size_t)blockIdx.x * 256 + threadIdx.x;
  if (i >= total) return;
  size_t r = i / Ncol, c = i % Ncol;
  out[c * (size_t)R + r] = to_bf16(in[i]);
}

// --------------------------------------------------------------- layernorm --
__global__ __launch_bounds__(256) void ln_kernel(
    const float* __restrict__ xin, const float* __restrict__ g,
    const float* __restrict__ bb, bf16* __restrict__ y) {
  int row = blockIdx.x;
  const float* xr = xin + (size_t)row * kC;
  __shared__ float red[256];
  int tid = threadIdx.x;

  float s = 0.f;
  for (int i = tid; i < kC; i += 256) s += xr[i];
  red[tid] = s; __syncthreads();
  for (int off = 128; off > 0; off >>= 1) {
    if (tid < off) red[tid] += red[tid + off];
    __syncthreads();
  }
  float mu = red[0] * (1.0f / kC);
  __syncthreads();

  float v = 0.f;
  for (int i = tid; i < kC; i += 256) { float d = xr[i] - mu; v += d * d; }
  red[tid] = v; __syncthreads();
  for (int off = 128; off > 0; off >>= 1) {
    if (tid < off) red[tid] += red[tid + off];
    __syncthreads();
  }
  float rstd = __frsqrt_rn(red[0] * (1.0f / kC) + 1e-5f);

  bf16* yr = y + (size_t)row * kC;
  for (int i = tid; i < kC; i += 256)
    yr[i] = to_bf16((xr[i] - mu) * rstd * g[i] + bb[i]);
}

// ===================== cooperative double-buffered GEMM pipeline =============
// Block tile 128(M) x 64(N); 8 waves, wave w owns rows [w*16, w*16+16) x 64.
// A/B tiles staged in LDS via async copies, ping-pong double buffer.

struct CoopLds {
  bf16 At[2][128 * 32];   // 8KB per buffer
  bf16 Bt[2][64 * 32];    // 4KB per buffer
};

__device__ __forceinline__ void coop_issue(
    const bf16* A, const bf16* BT, int K, int m0, int n0,
    unsigned ldsA, unsigned ldsB, int tid, int buf, int k0) {
  int ar = tid >> 2, kc = tid & 3;
  unsigned kb = (unsigned)(k0 * 2);
  unsigned la = ldsA + (unsigned)(buf * 128 * 32 * 2);
  unsigned lb = ldsB + (unsigned)(buf * 64 * 32 * 2);
  async_b128(la + (unsigned)(ar * 64 + kc * 16), A,
             (unsigned)(((size_t)(m0 + ar) * K + kc * 8) * 2) + kb);
  async_b128(la + (unsigned)((ar + 64) * 64 + kc * 16), A,
             (unsigned)(((size_t)(m0 + ar + 64) * K + kc * 8) * 2) + kb);
  async_b128(lb + (unsigned)(ar * 64 + kc * 16), BT,
             (unsigned)(((size_t)(n0 + ar) * K + kc * 8) * 2) + kb);
}

// K pipeline with 4 named accumulators (distinct B fragments -> clauses of
// ds_loads followed by 4 back-to-back WMMAs, no per-WMMA dscnt serialization).
__device__ __forceinline__ void coop_mainloop(
    CoopLds& sh, const bf16* A, const bf16* BT, int K,
    int m0, int n0, int wv, int lane, int tid,
    v8f& acc0, v8f& acc1, v8f& acc2, v8f& acc3) {
  unsigned ldsA = lds_addr(&sh.At[0][0]);
  unsigned ldsB = lds_addr(&sh.Bt[0][0]);
  int nk = K >> 5;
  coop_issue(A, BT, K, m0, n0, ldsA, ldsB, tid, 0, 0);
  for (int i = 0; i < nk; ++i) {
    wait_async0();          // this wave's loads for buf[i&1] complete
    __syncthreads();        // all waves' loads visible; prev reads done
    if (i + 1 < nk)
      coop_issue(A, BT, K, m0, n0, ldsA, ldsB, tid, (i + 1) & 1, (i + 1) * 32);
    int cur = i & 1;
    const bf16* Ab = &sh.At[cur][0];
    const bf16* Bb = &sh.Bt[cur][0];
    // load all fragments first (distinct regs), then 4 WMMAs back-to-back
    v16bf af  = load_a(Ab + (wv * 16 + (lane & 15)) * 32, 0, lane);
    v16bf bf0 = load_b(Bb + (0  + (lane & 15)) * 32, 0, lane);
    v16bf bf1 = load_b(Bb + (16 + (lane & 15)) * 32, 0, lane);
    v16bf bf2 = load_b(Bb + (32 + (lane & 15)) * 32, 0, lane);
    v16bf bf3 = load_b(Bb + (48 + (lane & 15)) * 32, 0, lane);
    acc0 = wmma_bf16(af, bf0, acc0);
    acc1 = wmma_bf16(af, bf1, acc1);
    acc2 = wmma_bf16(af, bf2, acc2);
    acc3 = wmma_bf16(af, bf3, acc3);
  }
}

// ----------------------------------------------------- generic coop GEMM ----
template <bool RELU, bool OUT_F32>
__global__ __launch_bounds__(256) void gemm_coop_kernel(
    const bf16* __restrict__ A, const bf16* __restrict__ BT,
    const float* __restrict__ bias, const float* __restrict__ resid,
    void* __restrict__ outv, int M, int N, int K) {
  __shared__ __attribute__((aligned(16))) CoopLds sh;
  int tid = threadIdx.x, lane = tid & 31, wv = tid >> 5;
  int nblk = N >> 6;
  int m0 = (blockIdx.x / nblk) * 128;
  int n0 = (blockIdx.x % nblk) * 64;

  v8f acc0 = {}, acc1 = {}, acc2 = {}, acc3 = {};
  coop_mainloop(sh, A, BT, K, m0, n0, wv, lane, tid, acc0, acc1, acc2, acc3);

  int rbase = m0 + wv * 16 + ((lane & 16) ? 8 : 0);
  v8f accs[4] = {acc0, acc1, acc2, acc3};
#pragma unroll
  for (int j = 0; j < 4; ++j) {
    int col = n0 + j * 16 + (lane & 15);
    float bv = bias ? bias[col] : 0.0f;
#pragma unroll
    for (int r = 0; r < 8; ++r) {
      size_t idx = (size_t)(rbase + r) * N + col;
      float v = accs[j][r] + bv;
      if (resid) v += resid[idx];
      if (RELU) v = v > 0.f ? v : 0.f;
      if (OUT_F32) ((float*)outv)[idx] = v;
      else         ((bf16*)outv)[idx] = to_bf16(v);
    }
  }
}

// ------------------------------------------------------------- QKV coop -----
// One block per (mat, b, h, 128-row slab). q,k -> [B,H,T,D]; v -> vT [B,H,D,T].
__global__ __launch_bounds__(256) void qkv_coop_kernel(
    const bf16* __restrict__ x0, const bf16* __restrict__ WqT,
    const bf16* __restrict__ WkT, const bf16* __restrict__ WvT,
    bf16* __restrict__ q, bf16* __restrict__ k, bf16* __restrict__ vT) {
  __shared__ __attribute__((aligned(16))) CoopLds sh;
  int tid = threadIdx.x, lane = tid & 31, wv = tid >> 5;
  int id = blockIdx.x;
  int mb  = id & 15;  id >>= 4;   // 2048/128 slabs
  int h   = id & 15;  id >>= 4;
  int b   = id & 3;   id >>= 2;
  int mat = id;                   // 0=q 1=k 2=v

  const bf16* A  = x0 + (size_t)b * kT * kC;
  const bf16* BT = (mat == 0 ? WqT : mat == 1 ? WkT : WvT) + (size_t)h * kD * kC;
  int m0 = mb * 128;

  v8f acc0 = {}, acc1 = {}, acc2 = {}, acc3 = {};
  coop_mainloop(sh, A, BT, kC, m0, 0, wv, lane, tid, acc0, acc1, acc2, acc3);

  size_t hb = (size_t)(b * kH + h);
  v8f accs[4] = {acc0, acc1, acc2, acc3};
  if (mat == 2) {           // transposed store: vT[b,h,d,t], 8 contiguous t
    int tcol = m0 + wv * 16 + ((lane & 16) ? 8 : 0);
#pragma unroll
    for (int j = 0; j < 4; ++j) {
      int drow = j * 16 + (lane & 15);
      v8bf pk;
#pragma unroll
      for (int r = 0; r < 8; ++r) pk[r] = to_bf16(accs[j][r]);
      *(v8bf*)(vT + hb * kD * kT + (size_t)drow * kT + tcol) = pk;
    }
  } else {
    bf16* out = (mat == 0 ? q : k) + hb * kT * kD;
    int rbase = m0 + wv * 16 + ((lane & 16) ? 8 : 0);
#pragma unroll
    for (int j = 0; j < 4; ++j) {
      int col = j * 16 + (lane & 15);
#pragma unroll
      for (int r = 0; r < 8; ++r)
        out[(size_t)(rbase + r) * kD + col] = to_bf16(accs[j][r]);
    }
  }
}

// ------------------------------------------------- flash attention (16 rows)
__global__ __launch_bounds__(256) void attn_kernel(
    const bf16* __restrict__ q, const bf16* __restrict__ k,
    const bf16* __restrict__ vT, bf16* __restrict__ ctx) {
  __shared__ __attribute__((aligned(16))) bf16 pbuf[8][16 * 32];
  int lane = threadIdx.x & 31;
  int wl = threadIdx.x >> 5;
  int w = blockIdx.x * 8 + wl;
  int tt = w & 127; w >>= 7;
  int h  = w & 15;  w >>= 4;
  int b  = w & 3;
  int t0 = tt << 4;

  size_t hb = (size_t)(b * kH + h);
  const bf16* qh = q  + hb * kT * kD;
  const bf16* kh = k  + hb * kT * kD;
  const bf16* vh = vT + hb * kD * kT;

  int arow  = lane & 15;
  int col   = lane & 15;
  int rbase = (lane & 16) ? 8 : 0;

  const bf16* qrow = qh + (size_t)(t0 + arow) * kD;
  v16bf a0 = load_a(qrow, 0, lane);
  v16bf a1 = load_a(qrow, 32, lane);

  float mrow[8], lrow[8];
  v8f acc0 = {}, acc1 = {}, acc2 = {}, acc3 = {};
#pragma unroll
  for (int r = 0; r < 8; ++r) { mrow[r] = -__builtin_inff(); lrow[r] = 0.f; }

  for (int s0 = 0; s0 <= t0 + 15; s0 += 32) {
    const bf16* k0r = kh + (size_t)(s0 + col) * kD;
    const bf16* k1r = kh + (size_t)(s0 + 16 + col) * kD;
    __builtin_prefetch(kh + (size_t)(s0 + 32 + col) * kD, 0, 3);
    // load the four K fragments first, then 4 WMMAs back-to-back
    v16bf kb0 = load_b(k0r, 0, lane);
    v16bf kb1 = load_b(k0r, 32, lane);
    v16bf kb2 = load_b(k1r, 0, lane);
    v16bf kb3 = load_b(k1r, 32, lane);
    v8f S0 = {}, S1 = {};
    S0 = wmma_bf16(a0, kb0, S0);
    S1 = wmma_bf16(a0, kb2, S1);
    S0 = wmma_bf16(a1, kb1, S0);
    S1 = wmma_bf16(a1, kb3, S1);

#pragma unroll
    for (int r = 0; r < 8; ++r) {
      int trow = t0 + rbase + r;
      float x0v = S0[r] * kScale; if (s0 + col > trow)      x0v = -1e30f;
      float x1v = S1[r] * kScale; if (s0 + 16 + col > trow) x1v = -1e30f;
      float mx = fmaxf(x0v, x1v);
      mx = fmaxf(mx, __shfl_xor(mx, 1, 16));
      mx = fmaxf(mx, __shfl_xor(mx, 2, 16));
      mx = fmaxf(mx, __shfl_xor(mx, 4, 16));
      mx = fmaxf(mx, __shfl_xor(mx, 8, 16));
      float mnew = fmaxf(mrow[r], mx);
      float fac  = __expf(mrow[r] - mnew);
      float p0   = __expf(x0v - mnew);
      float p1   = __expf(x1v - mnew);
      float sum = p0 + p1;
      sum += __shfl_xor(sum, 1, 16);
      sum += __shfl_xor(sum, 2, 16);
      sum += __shfl_xor(sum, 4, 16);
      sum += __shfl_xor(sum, 8, 16);
      lrow[r] = lrow[r] * fac + sum;
      mrow[r] = mnew;
      acc0[r] *= fac; acc1[r] *= fac; acc2[r] *= fac; acc3[r] *= fac;
      pbuf[wl][(rbase + r) * 32 + col]      = to_bf16(p0);
      pbuf[wl][(rbase + r) * 32 + 16 + col] = to_bf16(p1);
    }
    asm volatile("s_wait_dscnt 0x0" ::: "memory");  // LDS RAW within wave

    const bf16* pb = &pbuf[wl][0];
    v16bf pf = load_a(pb + arow * 32, 0, lane);
    v16bf vf0 = load_b(vh + (size_t)(col)      * kT, s0, lane);
    v16bf vf1 = load_b(vh + (size_t)(col + 16) * kT, s0, lane);
    v16bf vf2 = load_b(vh + (size_t)(col + 32) * kT, s0, lane);
    v16bf vf3 = load_b(vh + (size_t)(col + 48) * kT, s0, lane);
    acc0 = wmma_bf16(pf, vf0, acc0);
    acc1 = wmma_bf16(pf, vf1, acc1);
    acc2 = wmma_bf16(pf, vf2, acc2);
    acc3 = wmma_bf16(pf, vf3, acc3);
  }

  size_t obase = (size_t)b * kT * kC + (size_t)h * 64;
#pragma unroll
  for (int r = 0; r < 8; ++r) {
    float inv = 1.0f / lrow[r];
    size_t ro = obase + (size_t)(t0 + rbase + r) * kC + col;
    ctx[ro + 0]  = to_bf16(acc0[r] * inv);
    ctx[ro + 16] = to_bf16(acc1[r] * inv);
    ctx[ro + 32] = to_bf16(acc2[r] * inv);
    ctx[ro + 48] = to_bf16(acc3[r] * inv);
  }
}

// ------------------------------------------------------------------ launch --
extern "C" void kernel_launch(void* const* d_in, const int* in_sizes, int n_in,
                              void* d_out, int out_size, void* d_ws, size_t ws_size,
                              hipStream_t stream) {
  (void)in_sizes; (void)n_in; (void)out_size; (void)ws_size;
  const float* inputs = (const float*)d_in[0];
  const float* ln1_g  = (const float*)d_in[1];
  const float* ln1_b  = (const float*)d_in[2];
  const float* Wq     = (const float*)d_in[3];
  const float* Wk     = (const float*)d_in[4];
  const float* Wv     = (const float*)d_in[5];
  const float* Wp     = (const float*)d_in[6];
  const float* bp     = (const float*)d_in[7];
  const float* ln2_g  = (const float*)d_in[8];
  const float* ln2_b  = (const float*)d_in[9];
  const float* W1     = (const float*)d_in[10];
  const float* b1     = (const float*)d_in[11];
  const float* W2     = (const float*)d_in[12];
  const float* b2     = (const float*)d_in[13];

  char* ws = (char*)d_ws;
  size_t o = 0;
  auto take = [&](size_t elems, size_t esz) { char* p = ws + o; o += elems * esz; return p; };
  bf16* x0   = (bf16*)take((size_t)kBT * kC, 2);
  bf16* WqT  = (bf16*)take((size_t)kH * kD * kC, 2);
  bf16* WkT  = (bf16*)take((size_t)kH * kD * kC, 2);
  bf16* WvT  = (bf16*)take((size_t)kH * kD * kC, 2);
  bf16* WpT  = (bf16*)take((size_t)kC * kC, 2);
  bf16* W1T  = (bf16*)take((size_t)kF * kC, 2);
  bf16* W2T  = (bf16*)take((size_t)kC * kF, 2);
  bf16* qb   = (bf16*)take((size_t)kB * kH * kT * kD, 2);
  bf16* kb   = (bf16*)take((size_t)kB * kH * kT * kD, 2);
  bf16* vTb  = (bf16*)take((size_t)kB * kH * kD * kT, 2);
  bf16* ctx  = (bf16*)take((size_t)kBT * kC, 2);
  float* xr  = (float*)take((size_t)kBT * kC, 4);
  bf16* hb   = (bf16*)take((size_t)kBT * kC, 2);
  bf16* ff1  = (bf16*)take((size_t)kBT * kF, 2);

  cvt_t_kernel<<<dim3((kC * kD + 255) / 256, kH), 256, 0, stream>>>(Wq, WqT, kC, kD);
  cvt_t_kernel<<<dim3((kC * kD + 255) / 256, kH), 256, 0, stream>>>(Wk, WkT, kC, kD);
  cvt_t_kernel<<<dim3((kC * kD + 255) / 256, kH), 256, 0, stream>>>(Wv, WvT, kC, kD);
  cvt_t_kernel<<<dim3((kC * kC + 255) / 256, 1), 256, 0, stream>>>(Wp, WpT, kC, kC);
  cvt_t_kernel<<<dim3((kC * kF + 255) / 256, 1), 256, 0, stream>>>(W1, W1T, kC, kF);
  cvt_t_kernel<<<dim3((kF * kC + 255) / 256, 1), 256, 0, stream>>>(W2, W2T, kF, kC);

  ln_kernel<<<kBT, 256, 0, stream>>>(inputs, ln1_g, ln1_b, x0);

  qkv_coop_kernel<<<3 * kB * kH * (kT / 128), 256, 0, stream>>>(
      x0, WqT, WkT, WvT, qb, kb, vTb);

  attn_kernel<<<(kB * kH * (kT / 16)) / 8, 256, 0, stream>>>(qb, kb, vTb, ctx);

  gemm_coop_kernel<false, true>
      <<<(kBT / 128) * (kC / 64), 256, 0, stream>>>(
          ctx, WpT, bp, inputs, xr, kBT, kC, kC);

  ln_kernel<<<kBT, 256, 0, stream>>>(xr, ln2_g, ln2_b, hb);

  gemm_coop_kernel<true, false>
      <<<(kBT / 128) * (kF / 64), 256, 0, stream>>>(
          hb, W1T, b1, nullptr, ff1, kBT, kF, kC);

  gemm_coop_kernel<false, true>
      <<<(kBT / 128) * (kC / 64), 256, 0, stream>>>(
          ff1, W2T, b2, xr, (float*)d_out, kBT, kC, kF);
}
